// SwitchTransformerEncoderLayer_11871289606924
// MI455X (gfx1250) — matrix operand order repaired and checked by reading, not compiled, not run on hardware
//
#include <hip/hip_runtime.h>
#include <cstdint>

#define S_  1024
#define B_  2
#define D_  1024
#define H_  16
#define E_  8
#define F_  4096
#define HD_ 64
#define N_  (S_*B_)

typedef __attribute__((ext_vector_type(16))) __bf16 v16bf;
typedef __attribute__((ext_vector_type(8)))  float  v8f;

union AF { v16bf v; __bf16 e[16]; };
union BF { v16bf v; __bf16 e[16]; unsigned int u[8]; };

__device__ __forceinline__ float redmax16(float v) {
  v = fmaxf(v, __shfl_xor(v, 1, 32));
  v = fmaxf(v, __shfl_xor(v, 2, 32));
  v = fmaxf(v, __shfl_xor(v, 4, 32));
  v = fmaxf(v, __shfl_xor(v, 8, 32));
  return v;
}
__device__ __forceinline__ float redsum16(float v) {
  v += __shfl_xor(v, 1, 32);
  v += __shfl_xor(v, 2, 32);
  v += __shfl_xor(v, 4, 32);
  v += __shfl_xor(v, 8, 32);
  return v;
}

__device__ __forceinline__ unsigned pack_bf16(float a, float b) {
  union { unsigned u; __bf16 h[2]; } x;
  x.h[0] = (__bf16)a; x.h[1] = (__bf16)b;
  return x.u;
}

// Issue 4 x async 16B global->LDS DMA (64B total), tracked by ASYNCcnt.
// offset applies to BOTH global and LDS addresses (ISA 10.x async pseudocode),
// so one LDS base + one 64-bit global address covers 64 contiguous bytes.
#define ASYNC_LDS_B128_X4(ldsaddr, gaddr)                                     \
  asm volatile(                                                               \
      "global_load_async_to_lds_b128 %0, %1, off\n\t"                         \
      "global_load_async_to_lds_b128 %0, %1, off offset:16\n\t"               \
      "global_load_async_to_lds_b128 %0, %1, off offset:32\n\t"               \
      "global_load_async_to_lds_b128 %0, %1, off offset:48"                   \
      :: "v"(ldsaddr), "v"(gaddr) : "memory")

#define WAIT_ASYNC0() asm volatile("s_wait_asynccnt 0" ::: "memory")

__device__ __forceinline__ unsigned lds_addr(const void* p) {
  return (unsigned)(size_t)p;  // flat LDS addr low 32 bits = group-segment offset
}

// ---------------------------------------------------------------------------
// Generic tiled bf16-WMMA GEMM:  C[I,J] = op(A[I,K] x B) + bias
//   BT=true : B stored [J,K] row-major (K contiguous)  -> async fp32 staging
//   BT=false: B stored [K,J] row-major (K strided)     -> packed bf16 transpose
//   GATHER_A: A rows gathered via sorted token list (MoE w1), C stored at slots
//   SCATTER : A rows are slots, C scattered to token rows scaled by gate (w2)
// Block tile 128x128, 256 threads = 8 waves, each wave 32x64 (2x4 fragments).
// A tile always staged fp32 via GLOBAL_LOAD_ASYNC_TO_LDS_B128.
// ---------------------------------------------------------------------------
template<bool BT, bool RELU, bool GATHER_A, bool SCATTER>
__global__ __launch_bounds__(256)
void gemm_wmma(const float* __restrict__ A, int lda,
               const float* __restrict__ Bsrc, const float* __restrict__ bias,
               float* __restrict__ C, int ldc,
               int I_total, int K, int J,
               size_t bStrideE, int biasStrideE,
               const int* __restrict__ sorted, const int* __restrict__ offs,
               const int* __restrict__ counts, const float* __restrict__ gate)
{
  __shared__ float  As32[128][36];                 // fp32 A tile, 144B row stride
  __shared__ float  Bs32[BT ? 128 : 4][36];        // fp32 B tile (BT path)
  __shared__ __bf16 Bsbf[BT ? 4 : 128][40];        // bf16 B tile (NN path)

  const int t  = threadIdx.x;
  const int w  = t >> 5;
  const int l  = t & 31;
  const int l16 = l & 15;
  const int hi  = l >> 4;

  int rowCount = I_total, rowBase = 0;
  if (GATHER_A || SCATTER) { rowCount = counts[blockIdx.z]; rowBase = offs[blockIdx.z]; }
  const int i0 = blockIdx.x * 128;
  if (i0 >= rowCount) return;
  const int j0 = blockIdx.y * 128;

  const float* Bp    = Bsrc + (size_t)blockIdx.z * bStrideE;
  const float* biasp = bias + (size_t)blockIdx.z * biasStrideE;

  const int iw = (w & 3) * 32;   // wave tile row offset (4 waves along I)
  const int jw = (w >> 2) * 64;  // wave tile col offset (2 waves along J)

  // Per-thread source row for the cooperative A load (fixed across K-steps)
  const int a_row  = t >> 1;
  const int a_kseg = (t & 1) * 16;
  size_t arow;
  {
    const int gi = i0 + a_row;
    if (GATHER_A) {
      const int ii = (gi < rowCount) ? gi : (rowCount - 1);
      arow = (size_t)sorted[rowBase + ii];
    } else if (SCATTER) {
      const int ii = (gi < rowCount) ? gi : (rowCount - 1);
      arow = (size_t)(rowBase + ii);
    } else {
      arow = (size_t)gi;
    }
  }
  const unsigned a_lds = lds_addr(&As32[a_row][a_kseg]);

  const v8f vz = {0,0,0,0,0,0,0,0};
  v8f acc[2][4];
#pragma unroll
  for (int ti = 0; ti < 2; ++ti)
#pragma unroll
    for (int tj = 0; tj < 4; ++tj) acc[ti][tj] = vz;

  for (int k0 = 0; k0 < K; k0 += 32) {
    // ---- A tile (128 x 32 fp32): async DMA straight into LDS ----
    {
      const unsigned long long ga =
          (unsigned long long)(size_t)(A + arow * (size_t)lda + k0 + a_kseg);
      ASYNC_LDS_B128_X4(a_lds, ga);
    }
    // ---- B tile ----
    if (BT) {  // B[J,K], K contiguous: async fp32 staging
      const int j    = t >> 1;
      const int kseg = (t & 1) * 16;
      const unsigned long long gb =
          (unsigned long long)(size_t)(Bp + (size_t)(j0 + j) * K + k0 + kseg);
      ASYNC_LDS_B128_X4(lds_addr(&Bs32[j][kseg]), gb);
    } else {   // B[K,J], J contiguous: 4k x 4j sub-block, packed b32 transpose
      const int kq = (w) * 4;          // 8 waves x 4 k-rows = 32
      const int jb = (l) * 4;          // 32 lanes x 4 j = 128
      const float4 r0 = *(const float4*)(Bp + (size_t)(k0 + kq + 0) * J + j0 + jb);
      const float4 r1 = *(const float4*)(Bp + (size_t)(k0 + kq + 1) * J + j0 + jb);
      const float4 r2 = *(const float4*)(Bp + (size_t)(k0 + kq + 2) * J + j0 + jb);
      const float4 r3 = *(const float4*)(Bp + (size_t)(k0 + kq + 3) * J + j0 + jb);
      *(unsigned*)&Bsbf[jb + 0][kq]     = pack_bf16(r0.x, r1.x);
      *(unsigned*)&Bsbf[jb + 0][kq + 2] = pack_bf16(r2.x, r3.x);
      *(unsigned*)&Bsbf[jb + 1][kq]     = pack_bf16(r0.y, r1.y);
      *(unsigned*)&Bsbf[jb + 1][kq + 2] = pack_bf16(r2.y, r3.y);
      *(unsigned*)&Bsbf[jb + 2][kq]     = pack_bf16(r0.z, r1.z);
      *(unsigned*)&Bsbf[jb + 2][kq + 2] = pack_bf16(r2.z, r3.z);
      *(unsigned*)&Bsbf[jb + 3][kq]     = pack_bf16(r0.w, r1.w);
      *(unsigned*)&Bsbf[jb + 3][kq + 2] = pack_bf16(r2.w, r3.w);
    }
    WAIT_ASYNC0();
    __syncthreads();

    // ---- fragments in contiguous k order (k = 16*hi + s) (fp32 LDS -> bf16 at read) ----
    AF a[2];
#pragma unroll
    for (int ti = 0; ti < 2; ++ti) {
      const int row = iw + ti*16 + l16;
#pragma unroll
      for (int i = 0; i < 8; ++i) {
        const int kk = 2*i + 16*hi;
        const float2 v = *(const float2*)&As32[row][kk];
        a[ti].e[2*i]   = (__bf16)v.x;
        a[ti].e[2*i+1] = (__bf16)v.y;
      }
    }
    BF bfr[4];
#pragma unroll
    for (int tj = 0; tj < 4; ++tj) {
      const int col = jw + tj*16 + l16;
      if (BT) {
        const float4* p = (const float4*)&Bs32[col][16*hi];
#pragma unroll
        for (int q = 0; q < 4; ++q) {
          const float4 v = p[q];
          bfr[tj].e[4*q+0] = (__bf16)v.x; bfr[tj].e[4*q+1] = (__bf16)v.y;
          bfr[tj].e[4*q+2] = (__bf16)v.z; bfr[tj].e[4*q+3] = (__bf16)v.w;
        }
      } else {
        const unsigned int* p32 = (const unsigned int*)&Bsbf[col][16*hi];
#pragma unroll
        for (int q = 0; q < 8; ++q) bfr[tj].u[q] = p32[q];
      }
    }
#pragma unroll
    for (int ti = 0; ti < 2; ++ti)
#pragma unroll
      for (int tj = 0; tj < 4; ++tj)
        acc[ti][tj] = __builtin_amdgcn_wmma_f32_16x16x32_bf16(
            false, a[ti].v, false, bfr[tj].v, (short)0, acc[ti][tj], false, false);
    __syncthreads();
  }

  // ---- epilogue ----
#pragma unroll
  for (int ti = 0; ti < 2; ++ti) {
#pragma unroll
    for (int tj = 0; tj < 4; ++tj) {
      const int gj = j0 + jw + tj*16 + l16;
      const float bv = biasp[gj];
#pragma unroll
      for (int r = 0; r < 8; ++r) {
        const int gi = i0 + iw + ti*16 + r + 8*hi;
        float v = acc[ti][tj][r] + bv;
        if (RELU) v = fmaxf(v, 0.f);
        if (SCATTER) {
          if (gi < rowCount) {
            const int tok = sorted[rowBase + gi];
            C[(size_t)tok * ldc + gj] = gate[tok] * v;
          }
        } else if (GATHER_A) {
          if (gi < rowCount) C[(size_t)(rowBase + gi) * ldc + gj] = v;
        } else {
          C[(size_t)gi * ldc + gj] = v;
        }
      }
    }
  }
}

// ---------------------------------------------------------------------------
// Flash attention: grid (S/128, B*H), 8 waves/block, one 16-row q-tile per wave.
// qkv: [N, 3D] fp32.  ctx out: [N, D] fp32.
// ---------------------------------------------------------------------------
__global__ __launch_bounds__(256)
void attn_kernel(const float* __restrict__ qkv, float* __restrict__ ctx)
{
  __shared__ __bf16 Pst[8][16][40];  // per-wave P staging (16 x 32 + pad)
  __shared__ __bf16 Vt [8][64][40];  // per-wave transposed V chunk [d][k]

  const int t = threadIdx.x;
  const int w = t >> 5, l = t & 31, l16 = l & 15, hi = l >> 4;
  const int bh = blockIdx.y, b = bh & (B_ - 1), h = bh >> 1;
  const int q0 = (blockIdx.x * 8 + w) * 16;

  // Q as two A-fragments (d = 0..31, 32..63), pre-scaled by 1/sqrt(HD)
  AF qa[2];
  {
    const size_t n = (size_t)(q0 + l16) * B_ + b;
    const float* qp = qkv + n * (3*D_) + (size_t)h * HD_;
#pragma unroll
    for (int c = 0; c < 2; ++c)
#pragma unroll
      for (int i = 0; i < 8; ++i) {
        const int kk = 2*i + 16*hi + 32*c;
        float2 v = *(const float2*)(qp + kk);
        qa[c].e[2*i]   = (__bf16)(v.x * 0.125f);
        qa[c].e[2*i+1] = (__bf16)(v.y * 0.125f);
      }
  }

  const v8f vz = {0,0,0,0,0,0,0,0};
  v8f o[4]; o[0]=vz; o[1]=vz; o[2]=vz; o[3]=vz;
  float m[8], ss[8];
#pragma unroll
  for (int r = 0; r < 8; ++r) { m[r] = -1e30f; ss[r] = 0.f; }

  for (int kt = 0; kt < S_/32; ++kt) {
    const int kbase = kt * 32;

    // stage V chunk (32 x 64) transposed into Vt[w][d][k] (scalar bf16 stores)
    {
      const size_t nv = (size_t)(kbase + l) * B_ + b;
      const float* vp = qkv + nv * (3*D_) + 2*D_ + (size_t)h * HD_;
#pragma unroll
      for (int d4 = 0; d4 < 16; ++d4) {
        float4 v = *(const float4*)(vp + 4*d4);
        Vt[w][4*d4+0][l] = (__bf16)v.x;
        Vt[w][4*d4+1][l] = (__bf16)v.y;
        Vt[w][4*d4+2][l] = (__bf16)v.z;
        Vt[w][4*d4+3][l] = (__bf16)v.w;
      }
    }

    // scores for 32 k-columns: two 16x16 D-frags, each = 2 WMMAs over d
    v8f sc[2]; sc[0] = vz; sc[1] = vz;
#pragma unroll
    for (int g = 0; g < 2; ++g) {
      const size_t nk = (size_t)(kbase + g*16 + l16) * B_ + b;
      const float* kp = qkv + nk * (3*D_) + D_ + (size_t)h * HD_;
#pragma unroll
      for (int c = 0; c < 2; ++c) {
        BF kb;
        const float4* k4 = (const float4*)(kp + 32*c + 16*hi);
#pragma unroll
        for (int q = 0; q < 4; ++q) {
          float4 v = k4[q];
          kb.e[4*q+0] = (__bf16)v.x; kb.e[4*q+1] = (__bf16)v.y;
          kb.e[4*q+2] = (__bf16)v.z; kb.e[4*q+3] = (__bf16)v.w;
        }
        sc[g] = __builtin_amdgcn_wmma_f32_16x16x32_bf16(
            false, qa[c].v, false, kb.v, (short)0, sc[g], false, false);
      }
    }

    // online softmax over these 32 columns; stage P to LDS
#pragma unroll
    for (int r = 0; r < 8; ++r) {
      const float x0 = sc[0][r], x1 = sc[1][r];
      const float mx = redmax16(fmaxf(x0, x1));
      const float mn = fmaxf(m[r], mx);
      const float corr = __expf(m[r] - mn);
      const float p0 = __expf(x0 - mn);
      const float p1 = __expf(x1 - mn);
      const float rs = redsum16(p0 + p1);
      ss[r] = ss[r] * corr + rs;
      m[r] = mn;
#pragma unroll
      for (int f = 0; f < 4; ++f) o[f][r] *= corr;
      Pst[w][r + 8*hi][l16]      = (__bf16)p0;
      Pst[w][r + 8*hi][16 + l16] = (__bf16)p1;
    }
    __syncthreads();

    // P (A-frag) x V (B-frags): accumulate ctx
    AF pa;
#pragma unroll
    for (int i = 0; i < 8; ++i) {
      const int kk = 2*i + 16*hi;
      pa.e[2*i]   = Pst[w][l16][kk];
      pa.e[2*i+1] = Pst[w][l16][kk+1];
    }
#pragma unroll
    for (int f = 0; f < 4; ++f) {
      BF vb;
      const unsigned int* p32 = (const unsigned int*)&Vt[w][f*16 + l16][16*hi];
#pragma unroll
      for (int q = 0; q < 8; ++q) vb.u[q] = p32[q];
      o[f] = __builtin_amdgcn_wmma_f32_16x16x32_bf16(
          false, pa.v, false, vb.v, (short)0, o[f], false, false);
    }
    __syncthreads();
  }

#pragma unroll
  for (int r = 0; r < 8; ++r) {
    const float inv = 1.f / ss[r];
    const size_t n = (size_t)(q0 + r + 8*hi) * B_ + b;
#pragma unroll
    for (int f = 0; f < 4; ++f)
      ctx[n * D_ + (size_t)h * HD_ + f*16 + l16] = o[f][r] * inv;
  }
}

// ---------------------------------------------------------------------------
// LayerNorm with fused residual: out = LN(A + Bq) * g + be.  One block / token.
// ---------------------------------------------------------------------------
__global__ __launch_bounds__(256)
void ln_kernel(const float* __restrict__ A, const float* __restrict__ Bq,
               const float* __restrict__ g, const float* __restrict__ be,
               float* __restrict__ out)
{
  __shared__ float r1[256], r2[256];
  const int n = blockIdx.x, t = threadIdx.x;
  const float* a  = A  + (size_t)n * D_;
  const float* bb = Bq + (size_t)n * D_;
  float x[4]; float s = 0.f, s2 = 0.f;
#pragma unroll
  for (int q = 0; q < 4; ++q) {
    const int idx = t*4 + q;
    x[q] = a[idx] + bb[idx];
    s += x[q]; s2 += x[q]*x[q];
  }
  r1[t] = s; r2[t] = s2; __syncthreads();
  for (int st = 128; st > 0; st >>= 1) {
    if (t < st) { r1[t] += r1[t + st]; r2[t] += r2[t + st]; }
    __syncthreads();
  }
  const float mu  = r1[0] * (1.f / D_);
  const float var = r2[0] * (1.f / D_) - mu * mu;
  const float inv = rsqrtf(var + 1e-5f);
#pragma unroll
  for (int q = 0; q < 4; ++q) {
    const int idx = t*4 + q;
    out[(size_t)n * D_ + idx] = (x[q] - mu) * inv * g[idx] + be[idx];
  }
}

// ---------------------------------------------------------------------------
// Router: single block. Softmax top-1, deterministic counting sort by expert,
// exact lb_loss via tree reduction. No atomics -> fully deterministic.
// ---------------------------------------------------------------------------
__global__ __launch_bounds__(256)
void router_kernel(const float* __restrict__ x,
                   const float* __restrict__ rw, const float* __restrict__ rb,
                   float* __restrict__ gate, int* __restrict__ sorted,
                   int* __restrict__ counts, int* __restrict__ offs,
                   float* __restrict__ lb_out)
{
  __shared__ unsigned char sIdx[N_];
  __shared__ float sP[256][E_];
  __shared__ int   sC[256][E_];
  __shared__ int   sOffs[E_];
  const int t = threadIdx.x;
  float pSum[E_]; int cnt[E_];
#pragma unroll
  for (int e = 0; e < E_; ++e) { pSum[e] = 0.f; cnt[e] = 0; }

  for (int it = 0; it < N_/256; ++it) {
    const int n = it*256 + t;
    const float4* xp = (const float4*)(x + (size_t)n * D_);
    float lg[E_];
#pragma unroll
    for (int e = 0; e < E_; ++e) lg[e] = rb[e];
    for (int q = 0; q < D_/4; ++q) {
      const float4 xv = xp[q];
#pragma unroll
      for (int e = 0; e < E_; ++e) {
        const float4 wv = ((const float4*)(rw + (size_t)e * D_))[q];
        lg[e] += xv.x*wv.x + xv.y*wv.y + xv.z*wv.z + xv.w*wv.w;
      }
    }
    float mx = lg[0];
#pragma unroll
    for (int e = 1; e < E_; ++e) mx = fmaxf(mx, lg[e]);
    float p[E_]; float den = 0.f;
#pragma unroll
    for (int e = 0; e < E_; ++e) { p[e] = __expf(lg[e] - mx); den += p[e]; }
    const float invd = 1.f / den;
    float best = -1.f; int bi = 0;
#pragma unroll
    for (int e = 0; e < E_; ++e) {
      p[e] *= invd;
      if (p[e] > best) { best = p[e]; bi = e; }
      pSum[e] += p[e];
    }
    cnt[bi]++;
    gate[n] = best;
    sIdx[n] = (unsigned char)bi;
  }
#pragma unroll
  for (int e = 0; e < E_; ++e) { sP[t][e] = pSum[e]; sC[t][e] = cnt[e]; }
  __syncthreads();
  for (int st = 128; st > 0; st >>= 1) {
    if (t < st)
#pragma unroll
      for (int e = 0; e < E_; ++e) { sP[t][e] += sP[t+st][e]; sC[t][e] += sC[t+st][e]; }
    __syncthreads();
  }
  if (t == 0) {
    int run = 0; float lb = 0.f;
    for (int e = 0; e < E_; ++e) {
      const int c = sC[0][e];
      counts[e] = c; offs[e] = run; sOffs[e] = run; run += c;
      lb += ((float)c / N_) * (sP[0][e] / N_);
    }
    lb_out[0] = (float)E_ * lb;
  }
  __syncthreads();
  // deterministic rank within expert segment
  for (int it = 0; it < N_/256; ++it) {
    const int n = it*256 + t;
    const int e = sIdx[n];
    int rank = 0;
    for (int mm = 0; mm < n; ++mm) rank += (sIdx[mm] == e);
    sorted[sOffs[e] + rank] = n;
  }
}

// ---------------------------------------------------------------------------
extern "C" void kernel_launch(void* const* d_in, const int* in_sizes, int n_in,
                              void* d_out, int out_size, void* d_ws, size_t ws_size,
                              hipStream_t stream) {
  (void)in_sizes; (void)n_in; (void)out_size; (void)ws_size;
  const float* src        = (const float*)d_in[0];
  const float* in_proj_w  = (const float*)d_in[1];
  const float* in_proj_b  = (const float*)d_in[2];
  const float* out_proj_w = (const float*)d_in[3];
  const float* out_proj_b = (const float*)d_in[4];
  const float* router_w   = (const float*)d_in[5];
  const float* router_b   = (const float*)d_in[6];
  const float* w1         = (const float*)d_in[7];
  const float* b1         = (const float*)d_in[8];
  const float* w2         = (const float*)d_in[9];
  const float* b2         = (const float*)d_in[10];
  const float* ln1_g      = (const float*)d_in[11];
  const float* ln1_b      = (const float*)d_in[12];
  const float* ln2_g      = (const float*)d_in[13];
  const float* ln2_b      = (const float*)d_in[14];
  float* out = (float*)d_out;

  float* qkv  = (float*)d_ws;                     // [N, 3D]
  float* ctx  = qkv  + (size_t)N_ * 3 * D_;       // [N, D]
  float* src2 = ctx  + (size_t)N_ * D_;           // [N, D] attn out; reused as MoE out
  float* xb   = src2 + (size_t)N_ * D_;           // [N, D] after ln1
  float* hb   = xb   + (size_t)N_ * D_;           // [N, F] expert hidden (slot order)
  float* gate = hb   + (size_t)N_ * F_;           // [N]
  int* sorted = (int*)(gate + N_);                // [N]
  int* counts = sorted + N_;                      // [E]
  int* offs   = counts + E_;                      // [E]
  float* lb   = out + (size_t)N_ * D_;            // scalar at end of d_out

  // 1) QKV projection: [N,D] x [3D,D]^T
  gemm_wmma<true, false, false, false><<<dim3(N_/128, (3*D_)/128, 1), 256, 0, stream>>>(
      src, D_, in_proj_w, in_proj_b, qkv, 3*D_, N_, D_, 3*D_, 0, 0,
      nullptr, nullptr, nullptr, nullptr);

  // 2) Flash attention
  attn_kernel<<<dim3(S_/128, B_*H_), 256, 0, stream>>>(qkv, ctx);

  // 3) Output projection
  gemm_wmma<true, false, false, false><<<dim3(N_/128, D_/128, 1), 256, 0, stream>>>(
      ctx, D_, out_proj_w, out_proj_b, src2, D_, N_, D_, D_, 0, 0,
      nullptr, nullptr, nullptr, nullptr);

  // 4) ln1: x = LN(src + src2)
  ln_kernel<<<N_, 256, 0, stream>>>(src, src2, ln1_g, ln1_b, xb);

  // 5) Router + deterministic token sort + lb_loss
  router_kernel<<<1, 256, 0, stream>>>(xb, router_w, router_b, gate, sorted, counts, offs, lb);

  // 6) MoE w1 (gather tokens by expert, ReLU): h = relu(x_g @ w1[e] + b1[e])
  gemm_wmma<false, true, true, false><<<dim3(N_/128, F_/128, E_), 256, 0, stream>>>(
      xb, D_, w1, b1, hb, F_, N_, D_, F_, (size_t)D_ * F_, F_,
      sorted, offs, counts, nullptr);

  // 7) MoE w2 (gated scatter-combine): moe[tok] = gate[tok]*(h @ w2[e] + b2[e])
  gemm_wmma<false, false, false, true><<<dim3(N_/128, D_/128, E_), 256, 0, stream>>>(
      hb, F_, w2, b2, src2, D_, N_, F_, D_, (size_t)F_ * D_, D_,
      sorted, offs, counts, gate);

  // 8) ln2: out = LN(x + moe)
  ln_kernel<<<N_, 256, 0, stream>>>(xb, src2, ln2_g, ln2_b, out);
}
